// MultiHeadAttention_15908558864400
// MI455X (gfx1250) — matrix operand order repaired and checked by reading, not compiled, not run on hardware
//
#include <hip/hip_runtime.h>
#include <stdint.h>

// ---------------------------------------------------------------------------
// MHA for MI455X (gfx1250, wave32, WMMA). All GEMMs via v_wmma_f32_16x16x32_f16.
//   B=4, N=1024, D=1024, H=16, DK=DV=64
//   out:  (B,N,H*DV)  = 4M f32   | attn: (H*B,N,N) = 64M f32
// ---------------------------------------------------------------------------

typedef __attribute__((ext_vector_type(16))) _Float16 v16h;
typedef __attribute__((ext_vector_type(8)))  _Float16 v8h;
typedef __attribute__((ext_vector_type(8)))  float    v8f;
typedef __attribute__((ext_vector_type(4)))  float    v4f;

static __device__ __forceinline__ v8f wmma16(v16h a, v16h b, v8f c) {
    // 8 args: (neg_a, A, neg_b, B, c_mod, C, reuse_a, reuse_b)
    return __builtin_amdgcn_wmma_f32_16x16x32_f16(false, a, false, b, (short)0, c, false, false);
}
static __device__ __forceinline__ v16h cat8(v8h a, v8h b) {
    return __builtin_shufflevector(a, b, 0,1,2,3,4,5,6,7,8,9,10,11,12,13,14,15);
}
static __device__ __forceinline__ v8f vzero8() {
    return (v8f){0.f,0.f,0.f,0.f,0.f,0.f,0.f,0.f};
}

#define NB    4
#define NH    16
// workspace layout (offsets in _Float16 elements)
#define OFF_QF   0u
#define OFF_KF   4194304u
#define OFF_VF   8388608u
#define OFF_WT   12582912u   // [proj][h][c=64][d=1024] f16 (transposed weights)
#define OFF_QH   15728640u   // [h][b*N+n][64]
#define OFF_KH   19922944u   // [h][b*N+m][64]
#define OFF_VHT  24117248u   // [h][b][v=64][m=1024]  (transposed V head)

// ---------------------------------------------------------------------------
// K0a: fp32 -> fp16 conversion of q,k,v (halves projection-GEMM read traffic)
// ---------------------------------------------------------------------------
__global__ __launch_bounds__(256) void cvt_f16_kernel(
    const float* __restrict__ q, const float* __restrict__ k, const float* __restrict__ v,
    _Float16* __restrict__ qf, _Float16* __restrict__ kf, _Float16* __restrict__ vf)
{
    const int p = blockIdx.y;
    const float*   S = (p == 0) ? q : (p == 1) ? k : v;
    _Float16*      D = (p == 0) ? qf : (p == 1) ? kf : vf;
    size_t i0 = ((size_t)blockIdx.x * 256 + threadIdx.x) * 8;   // 2048*256*8 = 4M
    v4f x0 = *(const v4f*)(S + i0);
    v4f x1 = *(const v4f*)(S + i0 + 4);
    v8h o;
#pragma unroll
    for (int j = 0; j < 4; ++j) { o[j] = (_Float16)x0[j]; o[4 + j] = (_Float16)x1[j]; }
    *(v8h*)(D + i0) = o;
}

// ---------------------------------------------------------------------------
// K0b: weight transpose + fp16: WT[p][h][c][d] = W_p[h][d][c]
// ---------------------------------------------------------------------------
__global__ __launch_bounds__(256) void wt_f16_kernel(
    const float* __restrict__ wq, const float* __restrict__ wk, const float* __restrict__ wv,
    _Float16* __restrict__ WT)
{
    const int p = blockIdx.y;
    const float* W = (p == 0) ? wq : (p == 1) ? wk : wv;
    int idx = blockIdx.x * 256 + threadIdx.x;          // 0 .. 16*64*1024-1
    int h = idx >> 16;
    int c = (idx >> 10) & 63;
    int d = idx & 1023;
    WT[((size_t)p * NH + h) * 65536 + (size_t)c * 1024 + d] =
        (_Float16)W[(size_t)h * 65536 + (size_t)d * 64 + c];
}

// ---------------------------------------------------------------------------
// K1: per-head projections (M=4096, N=64, K=1024 per head/proj).
// grid = (16 row-blocks, 3 proj, 16 heads), 256 thr = 8 waves.
// Each wave owns TWO 16x64 row tiles (32 rows): every weight B-fragment feeds
// 2 WMMAs. Fragment loads are software-pipelined one k-step ahead so the
// scheduler can overlap global loads with v_wmma issue.
// ---------------------------------------------------------------------------
__global__ __launch_bounds__(256) void proj_wmma_kernel(
    const _Float16* __restrict__ qf, const _Float16* __restrict__ kf, const _Float16* __restrict__ vf,
    const _Float16* __restrict__ WT,
    _Float16* __restrict__ qh, _Float16* __restrict__ kh, _Float16* __restrict__ vhT)
{
    const int p = blockIdx.y, h = blockIdx.z;
    const int tid = threadIdx.x, lane = tid & 31, wid = tid >> 5;
    const int l16 = lane & 15, hi = lane >> 4;
    const int rows0 = blockIdx.x * 256 + wid * 32;     // 32 rows per wave

    const _Float16* X = (p == 0) ? qf : (p == 1) ? kf : vf;
    const _Float16* W = WT + ((size_t)p * NH + h) * 65536;        // [c][d]
    const _Float16* xr[2];
    xr[0] = X + (size_t)(rows0 + l16) * 1024;
    xr[1] = xr[0] + 16 * 1024;

    v8f acc[2][4];
#pragma unroll
    for (int rt = 0; rt < 2; ++rt)
#pragma unroll
        for (int nt = 0; nt < 4; ++nt) acc[rt][nt] = vzero8();

    // prologue: fragments for kt = 0
    v16h acur[2], bcur[4];
#pragma unroll
    for (int rt = 0; rt < 2; ++rt)
        acur[rt] = cat8(*(const v8h*)(xr[rt] + hi * 16), *(const v8h*)(xr[rt] + hi * 16 + 8));
#pragma unroll
    for (int nt = 0; nt < 4; ++nt)
        bcur[nt] = *(const v16h*)(W + (size_t)(nt * 16 + l16) * 1024 + hi * 16);

#pragma unroll 2
    for (int kt = 0; kt < 32; ++kt) {
        const int kn = ((kt + 1) & 31) * 32;           // next k-step (wraps: harmless reload)
        v16h anext[2], bnext[4];
#pragma unroll
        for (int rt = 0; rt < 2; ++rt)
            anext[rt] = cat8(*(const v8h*)(xr[rt] + kn + hi * 16),
                             *(const v8h*)(xr[rt] + kn + hi * 16 + 8));
#pragma unroll
        for (int nt = 0; nt < 4; ++nt)
            bnext[nt] = *(const v16h*)(W + (size_t)(nt * 16 + l16) * 1024 + kn + hi * 16);

#pragma unroll
        for (int rt = 0; rt < 2; ++rt)
#pragma unroll
            for (int nt = 0; nt < 4; ++nt)
                acc[rt][nt] = wmma16(acur[rt], bcur[nt], acc[rt][nt]);

#pragma unroll
        for (int rt = 0; rt < 2; ++rt) acur[rt] = anext[rt];
#pragma unroll
        for (int nt = 0; nt < 4; ++nt) bcur[nt] = bnext[nt];
    }

    if (p < 2) {
        _Float16* O = ((p == 0) ? qh : kh) + (size_t)h * (4096 * 64);
#pragma unroll
        for (int rt = 0; rt < 2; ++rt)
#pragma unroll
            for (int nt = 0; nt < 4; ++nt)
#pragma unroll
                for (int r = 0; r < 8; ++r)
                    O[(size_t)(rows0 + rt * 16 + r + 8 * hi) * 64 + nt * 16 + l16] =
                        (_Float16)acc[rt][nt][r];
    } else {
        // store V transposed: vhT[h][b][v][m], m contiguous -> future B-frags contiguous
        const int bi = rows0 >> 10;
        _Float16* O = vhT + ((size_t)h * NB + bi) * 65536;
#pragma unroll
        for (int rt = 0; rt < 2; ++rt) {
            const int m0 = (rows0 & 1023) + rt * 16 + 8 * hi;
#pragma unroll
            for (int nt = 0; nt < 4; ++nt) {
                v8h vv;
#pragma unroll
                for (int r = 0; r < 8; ++r) vv[r] = (_Float16)acc[rt][nt][r];
                *(v8h*)(O + (size_t)(nt * 16 + l16) * 1024 + m0) = vv;
            }
        }
    }
}

// ---------------------------------------------------------------------------
// K2: attention for one (h, b, 128-row block).
//   - kh (1024x64 f16 = 128KB) staged via GLOBAL_LOAD_ASYNC_TO_LDS_B128
//     (ASYNCcnt path, no VGPR round trip), CDNA5 320KB-LDS resident
//   - pass1: S = Q K^T via WMMA (LDS B-frags, pipelined), raw scores + row max
//   - pass2: exp(s-max) round trip through the L2-resident score block
//   - pass3: normalize + final attn (non-temporal b128 stores) while feeding
//            P-fragments into O = P V WMMAs (pipelined V/P loads)
// ---------------------------------------------------------------------------
__global__ __launch_bounds__(256) void attn_wmma_kernel(
    const _Float16* __restrict__ qh, const _Float16* __restrict__ kh,
    const _Float16* __restrict__ vhT,
    float* __restrict__ outO, float* __restrict__ outA)
{
    __shared__ _Float16 khLds[1024 * 64];          // 128 KB
    __shared__ float    invLds[8][16];

    const int rb = blockIdx.x, b = blockIdx.y, h = blockIdx.z;
    const int tid = threadIdx.x, lane = tid & 31, wid = tid >> 5;
    const int l16 = lane & 15, hi = lane >> 4;

    // ---- stage kh[h][b] into LDS with async 128-bit LDS-direct loads ----
    {
        const _Float16* khg = kh + ((size_t)h * 4096 + (size_t)b * 1024) * 64;
        const unsigned ldsBase = (unsigned)(uintptr_t)(&khLds[0]);
        for (int it = tid; it < 8192; it += 256) {
            unsigned ldsOff = ldsBase + (unsigned)it * 16u;
            uint64_t gaddr  = (uint64_t)(uintptr_t)(khg) + (uint64_t)it * 16u;
            asm volatile("global_load_async_to_lds_b128 %0, %1, off"
                         :: "v"(ldsOff), "v"(gaddr) : "memory");
        }
    }
    // ---- prefetch vhT[h][b] (128KB) toward L2/L0 while asyncs run ----
    {
        const char* vp = (const char*)(vhT + ((size_t)h * NB + b) * 65536);
        for (int it = tid; it < 1024; it += 256) __builtin_prefetch(vp + (size_t)it * 128, 0, 1);
    }
    asm volatile("s_wait_asynccnt 0x0" ::: "memory");   // this wave's LDS writes done
    __syncthreads();                                    // all waves' writes visible

    const int r0 = rb * 128 + wid * 16;            // this wave's 16 query rows
    // Q A-fragments (K=64 -> two 32-k steps), loaded once, reused for 64 col tiles
    const _Float16* qrow = qh + ((size_t)h * 4096 + (size_t)b * 1024 + r0 + l16) * 64;
    v16h aq[2];
#pragma unroll
    for (int ks = 0; ks < 2; ++ks)
        aq[ks] = cat8(*(const v8h*)(qrow + ks * 32 + hi * 16),
                      *(const v8h*)(qrow + ks * 32 + hi * 16 + 8));

    float* attnB = outA + (size_t)(h * NB + b) * (1024 * 1024);
    const float scale = 0.125f;                    // 1/sqrt(64)

    // ---------------- pass 1: scores + running row max (pipelined) ----------------
    float maxr[8];
#pragma unroll
    for (int r = 0; r < 8; ++r) maxr[r] = -3.4e38f;

    v16h bk0 = *(const v16h*)&khLds[(size_t)l16 * 64 + hi * 16];
    v16h bk1 = *(const v16h*)&khLds[(size_t)l16 * 64 + 32 + hi * 16];
#pragma unroll 2
    for (int ct = 0; ct < 64; ++ct) {
        const int cn = (ct + 1) & 63;
        v16h nb0 = *(const v16h*)&khLds[(size_t)(cn * 16 + l16) * 64 + hi * 16];
        v16h nb1 = *(const v16h*)&khLds[(size_t)(cn * 16 + l16) * 64 + 32 + hi * 16];

        v8f acc = vzero8();
        acc = wmma16(aq[0], bk0, acc);
        acc = wmma16(aq[1], bk1, acc);
#pragma unroll
        for (int r = 0; r < 8; ++r) {
            float s = acc[r] * scale;
            maxr[r] = fmaxf(maxr[r], s);
            attnB[(size_t)(r0 + r + 8 * hi) * 1024 + ct * 16 + l16] = s;   // raw score
        }
        bk0 = nb0; bk1 = nb1;
    }
#pragma unroll
    for (int r = 0; r < 8; ++r)
#pragma unroll
        for (int off = 1; off < 16; off <<= 1)
            maxr[r] = fmaxf(maxr[r], __shfl_xor(maxr[r], off, 32));

    // ---------------- pass 2: exp + row sums (L2-resident round trip) ----------------
    float sumr[8];
#pragma unroll
    for (int r = 0; r < 8; ++r) sumr[r] = 0.f;

    for (int ct = 0; ct < 64; ++ct) {
#pragma unroll
        for (int r = 0; r < 8; ++r) {
            size_t idx = (size_t)(r0 + r + 8 * hi) * 1024 + ct * 16 + l16;
            float pv = __expf(attnB[idx] - maxr[r]);
            sumr[r] += pv;
            attnB[idx] = pv;
        }
    }
#pragma unroll
    for (int r = 0; r < 8; ++r)
#pragma unroll
        for (int off = 1; off < 16; off <<= 1)
            sumr[r] += __shfl_xor(sumr[r], off, 32);

    if (l16 == 0) {
#pragma unroll
        for (int r = 0; r < 8; ++r) invLds[wid][r + 8 * hi] = 1.0f / sumr[r];
    }
    __syncthreads();
    const float myinv = invLds[wid][l16];          // 1/rowsum for row r0+l16

    // pass-2 stores must be visible to cross-lane loads in pass 3
    asm volatile("s_wait_storecnt 0x0" ::: "memory");

    // ---------------- pass 3: normalize+store attn, O = P @ V (pipelined) ----------------
    v8f accO[4];
#pragma unroll
    for (int nt = 0; nt < 4; ++nt) accO[nt] = vzero8();

    const _Float16* vbase = vhT + ((size_t)h * NB + b) * 65536;
    float* prowBase = attnB + (size_t)(r0 + l16) * 1024;

    v4f  c[4];
    v16h bv[4];
    {
        const int off = hi * 8;
        c[0] = *(const v4f*)(prowBase + off);      c[1] = *(const v4f*)(prowBase + off + 4);
        c[2] = *(const v4f*)(prowBase + off + 16); c[3] = *(const v4f*)(prowBase + off + 20);
#pragma unroll
        for (int nt = 0; nt < 4; ++nt)
            bv[nt] = cat8(*(const v8h*)(vbase + (size_t)(nt * 16 + l16) * 1024 + hi * 8), *(const v8h*)(vbase + (size_t)(nt * 16 + l16) * 1024 + hi * 8 + 16));
    }

#pragma unroll 2
    for (int kt = 0; kt < 32; ++kt) {
        const int ktn  = (kt + 1) & 31;
        const int offn = ktn * 32 + hi * 8;
        v4f  nc[4];
        v16h nbv[4];
        nc[0] = *(const v4f*)(prowBase + offn);      nc[1] = *(const v4f*)(prowBase + offn + 4);
        nc[2] = *(const v4f*)(prowBase + offn + 16); nc[3] = *(const v4f*)(prowBase + offn + 20);
#pragma unroll
        for (int nt = 0; nt < 4; ++nt)
            nbv[nt] = cat8(*(const v8h*)(vbase + (size_t)(nt * 16 + l16) * 1024 + ktn * 32 + hi * 8), *(const v8h*)(vbase + (size_t)(nt * 16 + l16) * 1024 + ktn * 32 + hi * 8 + 16));

        // normalize current chunk, write final attn (non-temporal 128-bit), build P frag
        const int off = kt * 32 + hi * 8;
        v4f p0 = c[0] * myinv, p1 = c[1] * myinv, p2 = c[2] * myinv, p3 = c[3] * myinv;
        __builtin_nontemporal_store(p0, (v4f*)(prowBase + off));
        __builtin_nontemporal_store(p1, (v4f*)(prowBase + off + 4));
        __builtin_nontemporal_store(p2, (v4f*)(prowBase + off + 16));
        __builtin_nontemporal_store(p3, (v4f*)(prowBase + off + 20));
        v16h pa;
#pragma unroll
        for (int j = 0; j < 4; ++j) {
            pa[j]      = (_Float16)p0[j];
            pa[4 + j]  = (_Float16)p1[j];
            pa[8 + j]  = (_Float16)p2[j];
            pa[12 + j] = (_Float16)p3[j];
        }
#pragma unroll
        for (int nt = 0; nt < 4; ++nt)
            accO[nt] = wmma16(pa, bv[nt], accO[nt]);

#pragma unroll
        for (int j = 0; j < 4; ++j) { c[j] = nc[j]; bv[j] = nbv[j]; }
    }

    // out[b][n][h*64 + c]
#pragma unroll
    for (int nt = 0; nt < 4; ++nt)
#pragma unroll
        for (int r = 0; r < 8; ++r)
            __builtin_nontemporal_store(
                accO[nt][r],
                outO + ((size_t)b * 1024 + r0 + r + 8 * hi) * 1024 + h * 64 + nt * 16 + l16);
}

// ---------------------------------------------------------------------------
extern "C" void kernel_launch(void* const* d_in, const int* in_sizes, int n_in,
                              void* d_out, int out_size, void* d_ws, size_t ws_size,
                              hipStream_t stream) {
    const float* q  = (const float*)d_in[0];
    const float* k  = (const float*)d_in[1];
    const float* v  = (const float*)d_in[2];
    const float* wq = (const float*)d_in[3];
    const float* wk = (const float*)d_in[4];
    const float* wv = (const float*)d_in[5];

    _Float16* ws  = (_Float16*)d_ws;
    _Float16* qf  = ws + OFF_QF;
    _Float16* kf  = ws + OFF_KF;
    _Float16* vf  = ws + OFF_VF;
    _Float16* WT  = ws + OFF_WT;
    _Float16* qh  = ws + OFF_QH;
    _Float16* kh  = ws + OFF_KH;
    _Float16* vhT = ws + OFF_VHT;

    float* outO = (float*)d_out;           // (B,N,H*DV) = 4M floats
    float* outA = outO + 4194304;          // (H*B,N,N)  = 64M floats

    cvt_f16_kernel  <<<dim3(2048, 3), 256, 0, stream>>>(q, k, v, qf, kf, vf);
    wt_f16_kernel   <<<dim3(4096, 3), 256, 0, stream>>>(wq, wk, wv, WT);
    proj_wmma_kernel<<<dim3(16, 3, 16), 256, 0, stream>>>(qf, kf, vf, WT, qh, kh, vhT);
    attn_wmma_kernel<<<dim3(8, 4, 16), 256, 0, stream>>>(qh, kh, vhT, outO, outA);
}
